// GAT_36361193128013
// MI455X (gfx1250) — matrix operand order repaired and checked by reading, not compiled, not run on hardware
//
#include <hip/hip_runtime.h>
#include <hip/hip_bf16.h>
#include <math.h>

typedef __attribute__((ext_vector_type(16))) _Float16 v16h;
typedef __attribute__((ext_vector_type(8)))  _Float16 v8h;
typedef __attribute__((ext_vector_type(8)))  float    v8f;

#define D_HID     64
#define NEG_SLOPE 0.2f
#define NGRAPHS   512

// ---------------------------------------------------------------- helpers
__device__ __forceinline__ void atomicMaxF(float* addr, float val) {
  // signed-int max for non-negative floats, unsigned-int min for negative.
  if (val >= 0.0f) atomicMax((int*)addr, __float_as_int(val));
  else             atomicMin((unsigned int*)addr, (unsigned int)__float_as_int(val));
}

// ---------------------------------------------------------------- f32 -> f16
__global__ void cvt_f32_f16(const float* __restrict__ in, _Float16* __restrict__ out, int n) {
  int i = blockIdx.x * blockDim.x + threadIdx.x;
  if (i < n) out[i] = (_Float16)in[i];
}

// ---------------------------------------------------------------- WMMA GEMM
// C[N,64] = A[N,K] (f16, row-major) x B[K,64] (f16, row-major), f32 accum.
// One block = 128 threads = 4 waves; wave w owns column tile n_base = 16*w.
// Block b owns rows [16*b, 16*b+16).  N % 16 == 0, K % 32 == 0.
__global__ __launch_bounds__(128) void gemm_wmma_f16(
    const _Float16* __restrict__ A, const _Float16* __restrict__ B,
    float* __restrict__ C, int K) {
  const int lane = threadIdx.x & 31;
  const int wave = threadIdx.x >> 5;
  const int rb   = blockIdx.x * 16;
  const int nb   = wave * 16;
  const int m    = lane & 15;
  const int kh   = lane >> 4;          // 0: K-halves {0..7},{16..23}; 1: {8..15},{24..31}

  v8f acc = {};
  for (int kb = 0; kb < K; kb += 32) {
    // A tile 16x32: lane m holds K = kb + kh*8 + {0..7} and kb + 16 + kh*8 + {0..7}
    const _Float16* arow = A + (size_t)(rb + m) * K + kb + kh * 8;
    v8h alo = *(const v8h*)(arow);
    v8h ahi = *(const v8h*)(arow + 16);
    v16h a = __builtin_shufflevector(alo, ahi, 0,1,2,3,4,5,6,7,8,9,10,11,12,13,14,15);
    // B tile 32x16: lane holds column N = nb + (lane & 15), K = kb + kh*8 + {0..7} and kb + 16 + kh*8 + {0..7}
    const _Float16* bcol = B + (size_t)kb * D_HID + nb + (lane & 15);
    v16h b;
    #pragma unroll
    for (int s = 0; s < 16; ++s) {
      int kk = (s < 8) ? (kh * 8 + s) : (16 + kh * 8 + (s - 8));
      b[s] = bcol[(size_t)kk * D_HID];
    }
    acc = __builtin_amdgcn_wmma_f32_16x16x32_f16(false, a, false, b,
                                                 (short)0, acc, false, false);
  }
  // C/D layout: VGPR r -> M = r + 8*kh ; lane -> N = lane & 15
  const int col = lane & 15;
  float* crow = C + (size_t)(rb + 8 * kh) * D_HID + nb + col;
  #pragma unroll
  for (int r = 0; r < 8; ++r) crow[(size_t)r * D_HID] = acc[r];
}

// ---------------------------------------------------------------- node prep
// Per (node, head): attention dots, init m=-inf, s=0, agg=0.
__global__ void att_init(const float* __restrict__ h,
                         const float* __restrict__ att_src,
                         const float* __restrict__ att_dst,
                         float* __restrict__ a_s, float* __restrict__ a_d,
                         float* __restrict__ m,  float* __restrict__ s,
                         float* __restrict__ agg, int N) {
  int idx = blockIdx.x * blockDim.x + threadIdx.x;
  if (idx >= N * 4) return;
  int n = idx >> 2, hd = idx & 3;
  const float* hp = h + (size_t)n * D_HID + hd * 16;
  const float* as = att_src + hd * 16;
  const float* ad = att_dst + hd * 16;
  float ds = 0.f, dd = 0.f;
  #pragma unroll
  for (int c = 0; c < 16; ++c) { float v = hp[c]; ds += v * as[c]; dd += v * ad[c]; }
  a_s[idx] = ds; a_d[idx] = dd;
  m[idx] = __int_as_float(0xFF800000); // -inf
  s[idx] = 0.f;
  float* ap = agg + (size_t)n * D_HID + hd * 16;
  #pragma unroll
  for (int c = 0; c < 16; ++c) ap[c] = 0.f;
}

// ---------------------------------------------------------------- edge passes
__device__ __forceinline__ void edge_ids(const int* ei, int E, int e, int& s, int& d) {
  if (e < E) { s = ei[e]; d = ei[E + e]; } else { s = d = e - E; } // self-loops
}

__global__ void edge_max(const int* __restrict__ ei, int E, int N,
                         const float* __restrict__ a_s, const float* __restrict__ a_d,
                         float* __restrict__ m) {
  int e = blockIdx.x * blockDim.x + threadIdx.x;
  if (e >= E + N) return;
  int s, d; edge_ids(ei, E, e, s, d);
  float4 as4 = ((const float4*)a_s)[s];
  float4 ad4 = ((const float4*)a_d)[d];
  float ev[4] = {as4.x + ad4.x, as4.y + ad4.y, as4.z + ad4.z, as4.w + ad4.w};
  #pragma unroll
  for (int hd = 0; hd < 4; ++hd) {
    float v = ev[hd]; v = v > 0.f ? v : NEG_SLOPE * v;
    atomicMaxF(&m[d * 4 + hd], v);
  }
}

__global__ void edge_sum(const int* __restrict__ ei, int E, int N,
                         const float* __restrict__ a_s, const float* __restrict__ a_d,
                         const float* __restrict__ m, float* __restrict__ s_out) {
  int e = blockIdx.x * blockDim.x + threadIdx.x;
  if (e >= E + N) return;
  int s, d; edge_ids(ei, E, e, s, d);
  float4 as4 = ((const float4*)a_s)[s];
  float4 ad4 = ((const float4*)a_d)[d];
  float4 m4  = ((const float4*)m)[d];
  float ev[4] = {as4.x + ad4.x, as4.y + ad4.y, as4.z + ad4.z, as4.w + ad4.w};
  float mv[4] = {m4.x, m4.y, m4.z, m4.w};
  #pragma unroll
  for (int hd = 0; hd < 4; ++hd) {
    float v = ev[hd]; v = v > 0.f ? v : NEG_SLOPE * v;
    atomicAdd(&s_out[d * 4 + hd], __expf(v - mv[hd]));
  }
}

__global__ void edge_scatter(const int* __restrict__ ei, int E, int N,
                             const float* __restrict__ a_s, const float* __restrict__ a_d,
                             const float* __restrict__ m, const float* __restrict__ ssum,
                             const float* __restrict__ h, float* __restrict__ agg) {
  int e = blockIdx.x * blockDim.x + threadIdx.x;
  if (e >= E + N) return;
  int s, d; edge_ids(ei, E, e, s, d);
  float4 as4 = ((const float4*)a_s)[s];
  float4 ad4 = ((const float4*)a_d)[d];
  float4 m4  = ((const float4*)m)[d];
  float4 s4  = ((const float4*)ssum)[d];
  float ev[4] = {as4.x + ad4.x, as4.y + ad4.y, as4.z + ad4.z, as4.w + ad4.w};
  float mv[4] = {m4.x, m4.y, m4.z, m4.w};
  float sv[4] = {s4.x, s4.y, s4.z, s4.w};
  float alpha[4];
  #pragma unroll
  for (int hd = 0; hd < 4; ++hd) {
    float v = ev[hd]; v = v > 0.f ? v : NEG_SLOPE * v;
    alpha[hd] = __expf(v - mv[hd]) / (sv[hd] + 1e-16f);
  }
  const float* hs = h   + (size_t)s * D_HID;
  float*       ag = agg + (size_t)d * D_HID;
  #pragma unroll
  for (int hd = 0; hd < 4; ++hd) {
    #pragma unroll
    for (int c = 0; c < 16; ++c)
      atomicAdd(&ag[hd * 16 + c], alpha[hd] * hs[hd * 16 + c]);
  }
}

// ---------------------------------------------------------------- activation
// In-place tanh(agg + bias); also emit f16 copy as next layer's GEMM input.
__global__ void node_update(float* __restrict__ agg, const float* __restrict__ bias,
                            _Float16* __restrict__ out16, int n64) {
  int i = blockIdx.x * blockDim.x + threadIdx.x;
  if (i >= n64) return;
  float v = tanhf(agg[i] + bias[i & (D_HID - 1)]);
  agg[i] = v;
  out16[i] = (_Float16)v;
}

__global__ void head_mean(const float* __restrict__ hact, float* __restrict__ node_mean, int N) {
  int i = blockIdx.x * blockDim.x + threadIdx.x;
  if (i >= N * 16) return;
  int n = i >> 4, c = i & 15;
  const float* hp = hact + (size_t)n * D_HID + c;
  node_mean[i] = 0.25f * (hp[0] + hp[16] + hp[32] + hp[48]);
}

// ---------------------------------------------------------------- pooling
__global__ void pool_init(float* __restrict__ cnt, float* __restrict__ gsum,
                          float* __restrict__ gmax) {
  int i = blockIdx.x * blockDim.x + threadIdx.x;
  if (i < NGRAPHS * 16) { gsum[i] = 0.f; gmax[i] = __int_as_float(0xFF800000); }
  if (i < NGRAPHS) cnt[i] = 0.f;
}

__global__ void pool_scatter(const float* __restrict__ node_mean, const int* __restrict__ batch,
                             float* __restrict__ cnt, float* __restrict__ gsum,
                             float* __restrict__ gmax, int N) {
  int n = blockIdx.x * blockDim.x + threadIdx.x;
  if (n >= N) return;
  int g = batch[n];
  atomicAdd(&cnt[g], 1.0f);
  const float* nm = node_mean + (size_t)n * 16;
  #pragma unroll
  for (int c = 0; c < 16; ++c) {
    float v = nm[c];
    atomicAdd(&gsum[g * 16 + c], v);
    atomicMaxF(&gmax[g * 16 + c], v);
  }
}

__global__ void final_out(const float* __restrict__ gmax, const float* __restrict__ gsum,
                          const float* __restrict__ cnt, const float* __restrict__ Wout,
                          const float* __restrict__ bout, float* __restrict__ out) {
  int g = blockIdx.x * blockDim.x + threadIdx.x;
  if (g >= NGRAPHS) return;
  float inv = 1.0f / fmaxf(cnt[g], 1.0f);
  float acc = bout[0];
  #pragma unroll
  for (int c = 0; c < 16; ++c) {
    float sm = gsum[g * 16 + c];
    acc += gmax[g * 16 + c] * Wout[c];
    acc += sm * inv          * Wout[16 + c];
    acc += sm                * Wout[32 + c];
  }
  out[g] = acc;
}

// ---------------------------------------------------------------- launcher
static inline unsigned gsz(long long n, int t) { return (unsigned)((n + t - 1) / t); }

extern "C" void kernel_launch(void* const* d_in, const int* in_sizes, int n_in,
                              void* d_out, int out_size, void* d_ws, size_t ws_size,
                              hipStream_t stream) {
  const float* x     = (const float*)d_in[0];
  const int*   ei    = (const int*)d_in[1];
  const int*   batch = (const int*)d_in[2];
  const float* W[3]  = {(const float*)d_in[3], (const float*)d_in[7],  (const float*)d_in[11]};
  const float* AS[3] = {(const float*)d_in[4], (const float*)d_in[8],  (const float*)d_in[12]};
  const float* AD[3] = {(const float*)d_in[5], (const float*)d_in[9],  (const float*)d_in[13]};
  const float* BI[3] = {(const float*)d_in[6], (const float*)d_in[10], (const float*)d_in[14]};
  const float* Wout  = (const float*)d_in[15];
  const float* bout  = (const float*)d_in[16];
  float* out = (float*)d_out;

  const int N = in_sizes[0] / 128;     // 50000
  const int E = in_sizes[1] / 2;       // 1600000
  const int ET = E + N;                // edges incl. self-loops

  // ---- workspace carve (256B aligned slabs)
  char* p = (char*)d_ws;
  auto carve = [&](size_t bytes) -> void* {
    void* r = (void*)p; p += (bytes + 255) & ~(size_t)255; return r;
  };
  _Float16* in16 = (_Float16*)carve((size_t)N * 128 * sizeof(_Float16));
  _Float16* w16  = (_Float16*)carve((size_t)128 * 64 * sizeof(_Float16));
  float* h    = (float*)carve((size_t)N * D_HID * sizeof(float));
  float* agg  = (float*)carve((size_t)N * D_HID * sizeof(float));
  float* a_s  = (float*)carve((size_t)N * 4 * sizeof(float));
  float* a_d  = (float*)carve((size_t)N * 4 * sizeof(float));
  float* mbuf = (float*)carve((size_t)N * 4 * sizeof(float));
  float* sbuf = (float*)carve((size_t)N * 4 * sizeof(float));
  float* nmean = (float*)carve((size_t)N * 16 * sizeof(float));
  float* cnt  = (float*)carve(NGRAPHS * sizeof(float));
  float* gsum = (float*)carve(NGRAPHS * 16 * sizeof(float));
  float* gmax = (float*)carve(NGRAPHS * 16 * sizeof(float));

  const int T = 256;

  // layer-1 input: x -> f16
  cvt_f32_f16<<<gsz((long long)N * 128, T), T, 0, stream>>>(x, in16, N * 128);

  int K = 128;
  for (int l = 0; l < 3; ++l) {
    cvt_f32_f16<<<gsz((long long)K * 64, T), T, 0, stream>>>(W[l], w16, K * 64);
    gemm_wmma_f16<<<N / 16, 128, 0, stream>>>(in16, w16, h, K);
    att_init<<<gsz((long long)N * 4, T), T, 0, stream>>>(h, AS[l], AD[l],
                                                         a_s, a_d, mbuf, sbuf, agg, N);
    edge_max    <<<gsz(ET, T), T, 0, stream>>>(ei, E, N, a_s, a_d, mbuf);
    edge_sum    <<<gsz(ET, T), T, 0, stream>>>(ei, E, N, a_s, a_d, mbuf, sbuf);
    edge_scatter<<<gsz(ET, T), T, 0, stream>>>(ei, E, N, a_s, a_d, mbuf, sbuf, h, agg);
    node_update <<<gsz((long long)N * D_HID, T), T, 0, stream>>>(agg, BI[l], in16,
                                                                 N * D_HID);
    K = 64;
  }

  head_mean  <<<gsz((long long)N * 16, T), T, 0, stream>>>(agg, nmean, N);
  pool_init  <<<gsz(NGRAPHS * 16, T), T, 0, stream>>>(cnt, gsum, gmax);
  pool_scatter<<<gsz(N, T), T, 0, stream>>>(nmean, batch, cnt, gsum, gmax, N);
  final_out  <<<gsz(NGRAPHS, T), T, 0, stream>>>(gmax, gsum, cnt, Wout, bout, out);
}
